// GNNGraphClassifier_29892972380781
// MI455X (gfx1250) — matrix operand order repaired and checked by reading, not compiled, not run on hardware
//
#include <hip/hip_runtime.h>
#include <hip/hip_bf16.h>
#include <hip/hip_fp16.h>

#define G 128
#define D 32
#define F 64
#define AA 8
#define NPTS 50000
#define NTILES (NPTS / 16)   // 3125, exact
#define AF 512               // AA*F
#define EPSF 1e-5f

typedef __attribute__((ext_vector_type(16))) _Float16 v16h;
typedef __attribute__((ext_vector_type(8)))  float    v8f;

// ---------- helpers ----------

__device__ inline void lds_fence() {
  // wave-local LDS RAW fence (writes then cross-lane reads within one wave)
  asm volatile("s_wait_dscnt 0x0" ::: "memory");
}

// async global->LDS (CDNA5): per-lane 16B, tracked by ASYNCcnt
__device__ inline void async_copy_b128(void* lds_dst, const void* g_src) {
  unsigned l = (unsigned)(uintptr_t)lds_dst;  // LDS aperture: low 32b = offset
  asm volatile("global_load_async_to_lds_b128 %0, %1, off"
               :
               : "v"(l), "v"(g_src)
               : "memory");
}
__device__ inline void async_wait0() {
  asm volatile("s_wait_asynccnt 0x0" ::: "memory");
}

// order-preserving float -> uint bijection (for exact atomic max)
__device__ inline unsigned fkey(float f) {
  unsigned u = __float_as_uint(f);
  return (u & 0x80000000u) ? ~u : (u | 0x80000000u);
}
__device__ inline float fdec(unsigned u) {
  u = (u & 0x80000000u) ? (u & 0x7fffffffu) : ~u;
  return __uint_as_float(u);
}

__device__ inline float wredsum(float v) {
#pragma unroll
  for (int m = 16; m > 0; m >>= 1) v += __shfl_xor(v, m, 32);
  return v;
}

// A-matrix (16xK f16) fragment gather from row-major LDS tile.
// ISA layout: lane L holds row m=L%16; lanes 0-15 K in {0..7,16..23}+k0,
// lanes 16-31 K in {8..15,24..31}+k0.
__device__ inline v16h a_frag_lds(const _Float16* T, int ld, int k0) {
  int lane = threadIdx.x & 31;
  int m = lane & 15, hi = lane >> 4;
  v16h a;
#pragma unroll
  for (int e = 0; e < 16; ++e) {
    int k = k0 + ((e >> 3) << 4) + (hi << 3) + (e & 7);
    a[e] = T[m * ld + k];
  }
  return a;
}

// B-matrix (32x16, K x N) fragment gathered from row-major f32 weights,
// converted to f16, zero-padded beyond ncol.  lanes 0-15: K in {0..7,16..23}+k0,
// lanes 16-31: K in {8..15,24..31}+k0; N = lane%16 + n0.
__device__ inline v16h b_frag_f32(const float* W, int ld, int ncol, int n0, int k0) {
  int lane = threadIdx.x & 31;
  int n = lane & 15;
  int kb = ((lane >> 4) << 3) + k0;
  v16h b;
#pragma unroll
  for (int e = 0; e < 16; ++e) {
    float v = ((n0 + n) < ncol) ? W[(kb + ((e >> 3) << 4) + (e & 7)) * ld + n0 + n] : 0.f;
    b[e] = (_Float16)v;
  }
  return b;
}

__device__ inline v8f wmma_f16(v16h a, v16h b, v8f c) {
  return __builtin_amdgcn_wmma_f32_16x16x32_f16(false, a, false, b, (short)0, c,
                                                false, false);
}

// ---------- kernels ----------

__global__ void k_init(float* counts, float* pooled) {
  int i = blockIdx.x * blockDim.x + threadIdx.x;
  if (i < G) counts[i] = 0.f;
  if (i < G * D) pooled[i] = 0.f;
}

__global__ void k_counts(const int* __restrict__ batch, float* __restrict__ counts) {
  int i = blockIdx.x * blockDim.x + threadIdx.x;
  if (i < NPTS) atomicAdd(&counts[batch[i]], 1.0f);
}

// wave = node, lane = channel d.  h = tanh(LN(x@Wi + bi))
__global__ void k_embed(const float* __restrict__ x, const float* __restrict__ Wi,
                        const float* __restrict__ bi, const float* __restrict__ lw,
                        const float* __restrict__ lb, float* __restrict__ h,
                        _Float16* __restrict__ hh) {
  int wave = (blockIdx.x * blockDim.x + threadIdx.x) >> 5;
  int d = threadIdx.x & 31;
  if (wave >= NPTS) return;
  float x0 = x[wave * 3 + 0], x1 = x[wave * 3 + 1], x2 = x[wave * 3 + 2];
  float v = bi[d] + x0 * Wi[d] + x1 * Wi[32 + d] + x2 * Wi[64 + d];
  float m = wredsum(v) * (1.f / 32.f);
  float c = v - m;
  float var = wredsum(c * c) * (1.f / 32.f);
  float t = tanhf(c / sqrtf(var + EPSF) * lw[d] + lb[d]);
  h[wave * 32 + d] = t;
  hh[wave * 32 + d] = (_Float16)t;
}

__global__ void k_zero_agg(float* sums, unsigned* maxk, float* red) {
  int i = blockIdx.x * blockDim.x + threadIdx.x;
  if (i < G * AF) { sums[i] = 0.f; maxk[i] = 0u; }
  if (i < 2) red[i] = 0.f;
}

struct SmemK2 {
  _Float16 Ah[16 * 32];   // h tile (f16)
  _Float16 xpH[16 * 64];  // xp tile (f16)
  float    xpF[16 * 64];  // xp tile (f32)
  float    attn[16 * 8];
  int      bat[16];
};

// per-wave 16-node tile: xp = h@Win+b (WMMA), attn = exp(-|xp@Wsc+b|) (WMMA),
// then segment scatter of w = attn (x) xp into sums/max (sorted-run flush).
__global__ __launch_bounds__(128) void k_bip(
    const _Float16* __restrict__ hh, const int* __restrict__ batch,
    const float* __restrict__ Win, const float* __restrict__ bin,
    const float* __restrict__ Wsc, const float* __restrict__ bsc,
    _Float16* __restrict__ xphg, float* __restrict__ sums,
    unsigned* __restrict__ maxk) {
  __shared__ SmemK2 sm[4];
  int wid = threadIdx.x >> 5, lane = threadIdx.x & 31;
  int nl = lane & 15, hi = lane >> 4;
  SmemK2& S = sm[wid];
  int tile = blockIdx.x * 4 + wid;
  if (tile >= NTILES) return;

  // kick off async stage of the h tile into LDS (overlaps weight gathers)
  {
    const uint4* src = (const uint4*)(hh + (size_t)tile * 512);
    uint4* dst = (uint4*)S.Ah;
    async_copy_b128(dst + lane, src + lane);
    async_copy_b128(dst + lane + 32, src + lane + 32);
  }
  if (lane < 16) S.bat[lane] = batch[tile * 16 + lane];

  // per-wave weight fragments (f32 -> f16 gather, one-time)
  v16h bW[4];
#pragma unroll
  for (int j = 0; j < 4; ++j) bW[j] = b_frag_f32(Win, 64, 64, j * 16, 0);
  v16h bS[2];
#pragma unroll
  for (int ks = 0; ks < 2; ++ks) bS[ks] = b_frag_f32(Wsc, 8, 8, 0, ks * 32);
  float binv[4];
#pragma unroll
  for (int j = 0; j < 4; ++j) binv[j] = bin[nl + 16 * j];
  float bscv = (nl < 8) ? bsc[nl] : 0.f;

  async_wait0();
  lds_fence();

  // xp = h @ Win  (16x32 @ 32x64 : 4 WMMAs)
  v16h af = a_frag_lds(S.Ah, 32, 0);
  v8f xpacc[4];
#pragma unroll
  for (int j = 0; j < 4; ++j) {
    v8f z = {};
    xpacc[j] = wmma_f16(af, bW[j], z);
  }
#pragma unroll
  for (int j = 0; j < 4; ++j) {
#pragma unroll
    for (int r = 0; r < 8; ++r) {
      int m = r + 8 * hi, n = nl + 16 * j;
      float v = xpacc[j][r] + binv[j];
      S.xpF[m * 64 + n] = v;
      S.xpH[m * 64 + n] = (_Float16)v;
    }
  }
  lds_fence();
  // coalesced store of xp(f16) tile to global for the output GEMM
  {
    const uint4* src = (const uint4*)S.xpH;
    uint4* dst = (uint4*)(xphg + (size_t)tile * 1024);
#pragma unroll
    for (int q = 0; q < 4; ++q) dst[lane + 32 * q] = src[lane + 32 * q];
  }
  // attn = exp(-|xp @ Wsc + b|)  (16x64 @ 64x16pad : 2 WMMAs)
  v8f aacc = {};
#pragma unroll
  for (int ks = 0; ks < 2; ++ks) {
    v16h ax = a_frag_lds(S.xpH, 64, ks * 32);
    aacc = wmma_f16(ax, bS[ks], aacc);
  }
  if (nl < 8) {
#pragma unroll
    for (int r = 0; r < 8; ++r) {
      int m = r + 8 * hi;
      S.attn[m * 8 + nl] = __expf(-fabsf(aacc[r] + bscv));
    }
  }
  lds_fence();

  // segment scatter: lane owns 16 of the 512 (a,f) elements (stride-32)
  float acc[16], mx[16];
  int curg = -1;
#pragma unroll 1
  for (int i = 0; i < 16; ++i) {
    int g = S.bat[i];
    if (g != curg) {
      if (curg >= 0) {
#pragma unroll
        for (int j = 0; j < 16; ++j) {
          int idx = lane + 32 * j;
          atomicAdd(&sums[curg * AF + idx], acc[j]);
          atomicMax(&maxk[curg * AF + idx], fkey(mx[j]));
        }
      }
      curg = g;
#pragma unroll
      for (int j = 0; j < 16; ++j) { acc[j] = 0.f; mx[j] = -3.4e38f; }
    }
#pragma unroll
    for (int j = 0; j < 16; ++j) {
      int idx = lane + 32 * j;
      float w = S.attn[i * 8 + (idx >> 6)] * S.xpF[i * 64 + (idx & 63)];
      acc[j] += w;
      mx[j] = fmaxf(mx[j], w);
    }
  }
#pragma unroll
  for (int j = 0; j < 16; ++j) {
    int idx = lane + 32 * j;
    atomicAdd(&sums[curg * AF + idx], acc[j]);
    atomicMax(&maxk[curg * AF + idx], fkey(mx[j]));
  }
}

// per-graph GEMM: agg(G x 1024) @ Wout[96:](1024 x 32) -> aggproj(G x 32)
// 16 waves: (mtile 0..7) x (ntile 0..1), 32 K-steps each.
__global__ __launch_bounds__(512) void k_aggproj(
    const float* __restrict__ sums, const unsigned* __restrict__ maxk,
    const float* __restrict__ counts, const float* __restrict__ Wout,
    float* __restrict__ aggproj) {
  int wid = threadIdx.x >> 5, lane = threadIdx.x & 31;
  int mtile = wid >> 1, ntile = wid & 1;
  int nl = lane & 15, hi = lane >> 4;
  int g = mtile * 16 + nl;  // A-matrix row for this lane
  float cnt = counts[g];
  float rc = 1.f / fmaxf(cnt, 1.f);
  bool nz = cnt > 0.f;
  v8f acc = {};
  for (int ks = 0; ks < 32; ++ks) {
    int k0 = ks * 32;
    v16h a;
#pragma unroll
    for (int e = 0; e < 16; ++e) {
      int k = k0 + ((e >> 3) << 4) + (hi << 3) + (e & 7);
      int ai = k >> 7, r = k & 127;  // agg[g, ai*128 + r]: r<64 mean else max
      float v = 0.f;
      if (nz) {
        int base = g * AF + ai * 64;
        v = (r < 64) ? sums[base + r] * rc : fdec(maxk[base + r - 64]);
      }
      a[e] = (_Float16)v;
    }
    v16h b = b_frag_f32(Wout + 96 * 32, 32, 32, ntile * 16, k0);
    acc = wmma_f16(a, b, acc);
  }
#pragma unroll
  for (int r = 0; r < 8; ++r) {
    int gm = mtile * 16 + r + 8 * hi;
    aggproj[gm * 32 + ntile * 16 + nl] = acc[r];
  }
}

struct SmemK4 {
  _Float16 Ah[16 * 32];
  _Float16 Xp[16 * 64];
  int      bat[16];
};

// hnew = tanh(h@WoutA + xp@WoutB + aggproj[batch] + bout); also global
// sum / sum-of-squares for the graph-wide layernorm.
__global__ __launch_bounds__(128) void k_update(
    const _Float16* __restrict__ hh, const _Float16* __restrict__ xphg,
    const int* __restrict__ batch, const float* __restrict__ Wout,
    const float* __restrict__ bout, const float* __restrict__ aggproj,
    float* __restrict__ hout, float* __restrict__ red) {
  __shared__ SmemK4 sm[4];
  int wid = threadIdx.x >> 5, lane = threadIdx.x & 31;
  int nl = lane & 15, hi = lane >> 4;
  SmemK4& S = sm[wid];
  int tile = blockIdx.x * 4 + wid;
  float lsum = 0.f, lsum2 = 0.f;
  if (tile < NTILES) {
    // async stage h tile + xp tile into LDS, overlapping weight gathers
    {
      const uint4* s1 = (const uint4*)(hh + (size_t)tile * 512);
      uint4* d1 = (uint4*)S.Ah;
      async_copy_b128(d1 + lane, s1 + lane);
      async_copy_b128(d1 + lane + 32, s1 + lane + 32);
      const uint4* s2 = (const uint4*)(xphg + (size_t)tile * 1024);
      uint4* d2 = (uint4*)S.Xp;
#pragma unroll
      for (int q = 0; q < 4; ++q) async_copy_b128(d2 + lane + 32 * q, s2 + lane + 32 * q);
      if (lane < 16) S.bat[lane] = batch[tile * 16 + lane];
    }
    v16h bh[2], bx[2][2];
#pragma unroll
    for (int nt = 0; nt < 2; ++nt) {
      bh[nt] = b_frag_f32(Wout, 32, 32, nt * 16, 0);
#pragma unroll
      for (int ks = 0; ks < 2; ++ks)
        bx[nt][ks] = b_frag_f32(Wout + 32 * 32, 32, 32, nt * 16, ks * 32);
    }
    float boutv[2] = {bout[nl], bout[nl + 16]};
    async_wait0();
    lds_fence();
    v16h ah = a_frag_lds(S.Ah, 32, 0);
    v16h ax0 = a_frag_lds(S.Xp, 64, 0);
    v16h ax1 = a_frag_lds(S.Xp, 64, 32);
#pragma unroll
    for (int nt = 0; nt < 2; ++nt) {
      v8f acc = {};
      acc = wmma_f16(ah, bh[nt], acc);
      acc = wmma_f16(ax0, bx[nt][0], acc);
      acc = wmma_f16(ax1, bx[nt][1], acc);
#pragma unroll
      for (int r = 0; r < 8; ++r) {
        int m = r + 8 * hi, n = nl + nt * 16;
        int g = S.bat[m];
        float t = tanhf(acc[r] + aggproj[g * 32 + n] + boutv[nt]);
        hout[((size_t)tile * 16 + m) * 32 + n] = t;
        lsum += t;
        lsum2 += t * t;
      }
    }
  }
  lsum = wredsum(lsum);
  lsum2 = wredsum(lsum2);
  if (lane == 0) {
    atomicAdd(&red[0], lsum);
    atomicAdd(&red[1], lsum2);
  }
}

// graph-wide LN over all N*D elements (mean/std from red)
__global__ void k_gln(const float* __restrict__ red, const float* __restrict__ gw,
                      const float* __restrict__ gb, float* __restrict__ h,
                      _Float16* __restrict__ hh) {
  int i = blockIdx.x * blockDim.x + threadIdx.x;
  if (i >= NPTS * D) return;
  const float nd = 1.f / (float)(NPTS * D);
  float mean = red[0] * nd;
  float var = fmaxf(red[1] * nd - mean * mean, 0.f);
  float inv = 1.f / (sqrtf(var) + EPSF);
  int d = i & 31;
  float t = (h[i] - mean) * inv * gw[d] + gb[d];
  h[i] = t;
  hh[i] = (_Float16)t;
}

__global__ void k_pool(const float* __restrict__ h, const int* __restrict__ batch,
                       float* __restrict__ pooled) {
  int i = blockIdx.x * blockDim.x + threadIdx.x;
  if (i >= NPTS * D) return;
  atomicAdd(&pooled[batch[i >> 5] * 32 + (i & 31)], h[i]);
}

// wave = graph, lane = channel: 3x (matvec via shfl, LN, tanh), final dot.
__global__ void k_head(const float* __restrict__ pooled, const float* __restrict__ counts,
                       const float* __restrict__ Wp, const float* __restrict__ bp,
                       const float* __restrict__ lw, const float* __restrict__ lb,
                       const float* __restrict__ Wpo, const float* __restrict__ bpo,
                       float* __restrict__ out) {
  int wave = (blockIdx.x * blockDim.x + threadIdx.x) >> 5;
  int d = threadIdx.x & 31;
  if (wave >= G) return;
  float s = pooled[wave * 32 + d] / fmaxf(counts[wave], 1.f);
#pragma unroll 1
  for (int j = 0; j < 3; ++j) {
    float y = bp[j * 32 + d];
    for (int k = 0; k < 32; ++k) {
      float sk = __shfl(s, k, 32);
      y += sk * Wp[j * 1024 + k * 32 + d];
    }
    float m = wredsum(y) * (1.f / 32.f);
    float c = y - m;
    float var = wredsum(c * c) * (1.f / 32.f);
    s = tanhf(c / sqrtf(var + EPSF) * lw[j * 32 + d] + lb[j * 32 + d]);
  }
  float r = wredsum(s * Wpo[d]);
  if (d == 0) out[wave] = r + bpo[0];
}

// ---------- host ----------

extern "C" void kernel_launch(void* const* d_in, const int* in_sizes, int n_in,
                              void* d_out, int out_size, void* d_ws, size_t ws_size,
                              hipStream_t stream) {
  (void)in_sizes; (void)n_in; (void)out_size; (void)ws_size;
  const float* x     = (const float*)d_in[0];
  const int*   batch = (const int*)d_in[1];
  const float* Wi    = (const float*)d_in[2];
  const float* bi    = (const float*)d_in[3];
  const float* lni_w = (const float*)d_in[4];
  const float* lni_b = (const float*)d_in[5];
  const float* W_in  = (const float*)d_in[6];
  const float* b_in  = (const float*)d_in[7];
  const float* W_sc  = (const float*)d_in[8];
  const float* b_sc  = (const float*)d_in[9];
  const float* W_out = (const float*)d_in[10];
  const float* b_out = (const float*)d_in[11];
  const float* gln_w = (const float*)d_in[12];
  const float* gln_b = (const float*)d_in[13];
  const float* Wp    = (const float*)d_in[14];
  const float* bp    = (const float*)d_in[15];
  const float* lnp_w = (const float*)d_in[16];
  const float* lnp_b = (const float*)d_in[17];
  const float* Wpo   = (const float*)d_in[18];
  const float* bpo   = (const float*)d_in[19];
  float* out = (float*)d_out;

  char* base = (char*)d_ws;
  size_t off = 0;
  auto alloc = [&](size_t bytes) -> void* {
    void* r = base + off;
    off += (bytes + 255) & ~(size_t)255;
    return r;
  };
  float*    counts = (float*)alloc(G * 4);
  float*    red    = (float*)alloc(2 * 4);
  float*    pooled = (float*)alloc(G * D * 4);
  float*    sums   = (float*)alloc((size_t)G * AF * 4);
  unsigned* maxk   = (unsigned*)alloc((size_t)G * AF * 4);
  float*    aggp   = (float*)alloc(G * D * 4);
  float*    h      = (float*)alloc((size_t)NPTS * D * 4);
  _Float16* hh     = (_Float16*)alloc((size_t)NPTS * D * 2);
  _Float16* xph    = (_Float16*)alloc((size_t)NPTS * F * 2);

  k_init<<<(G * D + 255) / 256, 256, 0, stream>>>(counts, pooled);
  k_counts<<<(NPTS + 255) / 256, 256, 0, stream>>>(batch, counts);
  k_embed<<<(NPTS * 32 + 127) / 128, 128, 0, stream>>>(x, Wi, bi, lni_w, lni_b, h, hh);

  const int bipBlocks = (NTILES + 3) / 4;
  for (int it = 0; it < 3; ++it) {
    for (int j = 0; j < 3; ++j) {
      const float* Winj  = W_in + j * D * F;
      const float* binj  = b_in + j * F;
      const float* Wscj  = W_sc + j * F * AA;
      const float* bscj  = b_sc + j * AA;
      const float* Woutj = W_out + (size_t)j * 1120 * 32;
      const float* boutj = b_out + j * D;
      k_zero_agg<<<(G * AF + 255) / 256, 256, 0, stream>>>(sums, maxk, red);
      k_bip<<<bipBlocks, 128, 0, stream>>>(hh, batch, Winj, binj, Wscj, bscj, xph,
                                           sums, maxk);
      k_aggproj<<<1, 512, 0, stream>>>(sums, maxk, counts, Woutj, aggp);
      k_update<<<bipBlocks, 128, 0, stream>>>(hh, xph, batch, Woutj, boutj, aggp, h,
                                              red);
      k_gln<<<(NPTS * D + 255) / 256, 256, 0, stream>>>(red, gln_w + j * D,
                                                        gln_b + j * D, h, hh);
    }
  }
  k_pool<<<(NPTS * D + 255) / 256, 256, 0, stream>>>(h, batch, pooled);
  k_head<<<(G * 32 + 127) / 128, 128, 0, stream>>>(pooled, counts, Wp, bp, lnp_w,
                                                   lnp_b, Wpo, bpo, out);
}
